// AudioTokenizer_90254442758304
// MI455X (gfx1250) — matrix operand order repaired and checked by reading, not compiled, not run on hardware
//
#include <hip/hip_runtime.h>
#include <hip/hip_bf16.h>
#include <stdint.h>

// ---------------------------------------------------------------------------
// Residual VQ encode for MI455X (gfx1250, wave32, WMMA bf16, async-LDS staging)
//   features : (8, 4096, 512) f32      codebooks : (4, 4096, 512) f32
//   out      : tokens (8,4,4096) i32  ++  quantized_sum (8,4096,512) f32
// dist(r, v) = ||r||^2 - 2 r.cb_v + ||cb_v||^2 ; ||r||^2 constant per row -> drop.
// ---------------------------------------------------------------------------

typedef __bf16 bf16_t;
typedef __attribute__((ext_vector_type(16))) __bf16 v16bf;
typedef __attribute__((ext_vector_type(8)))  float  v8f;

#define NCB        4
#define VCODES     4096
#define DDIM       512
#define BBATCH     8
#define LSEQ       4096
#define ROWS       (BBATCH * LSEQ)     // 32768
#define MTILE      64                  // rows per workgroup
#define KSTEPS     (DDIM / 32)         // 16 wmma K-steps per tile
#define VTILES     (VCODES / 16)       // 256
#define CHUNK      32                  // codes staged in LDS per iteration
#define NCHUNKS    (VCODES / CHUNK)    // 128
#define TOK_ELEMS  (BBATCH * NCB * LSEQ)         // 131072
#define PACKED_PER_VTILE 8192          // ushorts per 16-code vtile (16*32*16)
#define CHUNK_U4   2048                // uint4 per staged chunk (32KB / 16B)

__device__ __forceinline__ unsigned short f2bf_bits(float f) {
    bf16_t h = (bf16_t)f;
    return __builtin_bit_cast(unsigned short, h);
}

// -------------------- pre-pass 1: pack codebooks to bf16 B-fragment order ---
// Layout: packed[((cb*VTILES + vt)*KSTEPS + ks)*512 + lane*16 + j]
//   lane 0..15  -> column v = vt*16 + lane,     K = ks*32 + j        (j = 0..15)
//   lane 16..31 -> column v = vt*16 + lane-16,  K = ks*32 + 16 + j
__global__ void rvq_pack_cb(const float* __restrict__ cb,
                            unsigned short* __restrict__ packed) {
    unsigned int idx  = blockIdx.x * 256u + threadIdx.x;   // < 8,388,608
    unsigned int j    = idx & 15u;
    unsigned int lane = (idx >> 4) & 31u;
    unsigned int ks   = (idx >> 9) & 15u;
    unsigned int vt   = (idx >> 13) & 255u;
    unsigned int c    = idx >> 21;
    unsigned int v    = vt * 16u + (lane & 15u);
    unsigned int k    = ks * 32u + (lane >> 4) * 16u + j;
    float f = cb[((size_t)(c * VCODES + v)) * DDIM + k];
    packed[idx] = f2bf_bits(f);
}

// -------------------- pre-pass 2: ||cb_v||^2 -------------------------------
__global__ void rvq_cbsq(const float* __restrict__ cb, float* __restrict__ cbsq) {
    int row  = blockIdx.x * 8 + (threadIdx.x >> 5);   // (cb, v) flattened
    int lane = threadIdx.x & 31;
    const float* p = cb + (size_t)row * DDIM;
    float s = 0.0f;
    for (int k = lane; k < DDIM; k += 32) { float x = p[k]; s += x * x; }
    #pragma unroll
    for (int off = 16; off > 0; off >>= 1) s += __shfl_down(s, off, 32);
    if (lane == 0) cbsq[row] = s;
}

// Issue one 32KB chunk (8 x b128 per thread) via the CDNA5 async-to-LDS path.
// Tracked by ASYNCcnt; the copy engine overlaps with WMMA compute.
__device__ __forceinline__ void issue_chunk_async(const uint4* __restrict__ src,
                                                  unsigned ldsBufOff, int tid) {
    #pragma unroll
    for (int k = 0; k < 8; ++k) {
        int idx = tid + 256 * k;
        unsigned lds = ldsBufOff + (unsigned)idx * 16u;
        unsigned long long ga = (unsigned long long)(uintptr_t)(src + idx);
        asm volatile("global_load_async_to_lds_b128 %0, %1, off"
                     :: "v"(lds), "v"(ga) : "memory");
    }
}

// -------------------- main fused RVQ kernel --------------------------------
// 256 threads = 8 waves. wave -> (mSub = wave&3 picks 16-row A subtile,
// nHalf = wave>>2 picks one of the two 16-code vtiles in the staged chunk).
__global__ __launch_bounds__(256)
void rvq_main(const float* __restrict__ feat,
              const float* __restrict__ cb,
              const unsigned short* __restrict__ packedB,
              const float* __restrict__ cbsq,
              int* __restrict__ tokOut,
              float* __restrict__ qOut) {
    __shared__ unsigned short      resB[MTILE * DDIM];       // residual bf16 (64 KB)
    __shared__ unsigned short      bTile[2][CHUNK * DDIM];   // double-buffered B (64 KB)
    __shared__ unsigned long long  minPack[MTILE];           // (dist-key << 32) | v
    __shared__ int                 tokHist[NCB][MTILE];

    const int tid   = threadIdx.x;
    const int lane  = tid & 31;
    const int wave  = tid >> 5;
    const int mSub  = wave & 3;
    const int nHalf = wave >> 2;
    const int rowBase = blockIdx.x * MTILE;

    const unsigned bufOff[2] = {
        (unsigned)(uintptr_t)&bTile[0][0],   // low 32 bits of a shared generic
        (unsigned)(uintptr_t)&bTile[1][0] }; // pointer == LDS byte offset

    for (int step = 0; step < NCB; ++step) {
        const uint4* stepSrc = (const uint4*)packedB +
            (size_t)step * VTILES * (PACKED_PER_VTILE / 8);

        // prologue: start streaming chunk 0 while we rebuild the residual
        issue_chunk_async(stepSrc, bufOff[0], tid);

        // ---- rebuild bf16 residual in LDS: feat - sum of prior quantized ----
        {
            const float4* f4 = (const float4*)feat + (size_t)rowBase * (DDIM / 4);
            for (int i = tid; i < MTILE * (DDIM / 4); i += 256) {
                int row = i >> 7;            // DDIM/4 == 128
                int c4  = i & 127;
                float4 r = f4[row * 128 + c4];
                for (int s = 0; s < step; ++s) {
                    const float4 q = ((const float4*)cb)[
                        ((size_t)(s * VCODES + tokHist[s][row])) * 128 + c4];
                    r.x -= q.x; r.y -= q.y; r.z -= q.z; r.w -= q.w;
                }
                uint2 pk;
                pk.x = (unsigned)f2bf_bits(r.x) | ((unsigned)f2bf_bits(r.y) << 16);
                pk.y = (unsigned)f2bf_bits(r.z) | ((unsigned)f2bf_bits(r.w) << 16);
                *(uint2*)(resB + row * DDIM + c4 * 4) = pk;   // one ds_store_b64
            }
            if (tid < MTILE) minPack[tid] = ~0ull;
        }
        __syncthreads();

        unsigned long long rmin[8];
        #pragma unroll
        for (int r = 0; r < 8; ++r) rmin[r] = ~0ull;

        for (int chunk = 0; chunk < NCHUNKS; ++chunk) {
            // stream the NEXT chunk into the other buffer; drain this chunk's 8
            if (chunk + 1 < NCHUNKS) {
                issue_chunk_async(stepSrc + (size_t)(chunk + 1) * CHUNK_U4,
                                  bufOff[(chunk + 1) & 1], tid);
                asm volatile("s_wait_asynccnt 0x8" ::: "memory");
            } else {
                asm volatile("s_wait_asynccnt 0x0" ::: "memory");
            }
            __syncthreads();   // all threads' async loads for this buffer landed

            const unsigned short* bt = bTile[chunk & 1];
            const int vt    = chunk * 2 + nHalf;
            const int vcode = vt * 16 + (lane & 15);
            const float cbs = cbsq[step * VCODES + vcode];

            // two independent WMMA accumulation chains (K split 0..255 / 256..511)
            v8f acc0 = {}, acc1 = {};
            const int arow = mSub * 16 + (lane & 15);
            const int asel = (lane >> 4) ? 16 : 0;
            #pragma unroll
            for (int ks = 0; ks < 8; ++ks) {
                {   // chain 0: kstep = ks
                    int aoff = arow * DDIM + ks * 32 + asel;
                    union { v16bf v; uint4 q[2]; } A;
                    A.q[0] = *(const uint4*)(resB + aoff);
                    A.q[1] = *(const uint4*)(resB + aoff + 8);
                    int boff = ((nHalf * 16 + ks) * 32 + lane) * 16;
                    union { v16bf v; uint4 q[2]; } Bf;
                    Bf.q[0] = *(const uint4*)(bt + boff);
                    Bf.q[1] = *(const uint4*)(bt + boff + 8);
                    acc0 = __builtin_amdgcn_wmma_f32_16x16x32_bf16(
                            false, A.v, false, Bf.v, (short)0, acc0, false, false);
                }
                {   // chain 1: kstep = ks + 8
                    int aoff = arow * DDIM + (ks + 8) * 32 + asel;
                    union { v16bf v; uint4 q[2]; } A;
                    A.q[0] = *(const uint4*)(resB + aoff);
                    A.q[1] = *(const uint4*)(resB + aoff + 8);
                    int boff = ((nHalf * 16 + ks + 8) * 32 + lane) * 16;
                    union { v16bf v; uint4 q[2]; } Bf;
                    Bf.q[0] = *(const uint4*)(bt + boff);
                    Bf.q[1] = *(const uint4*)(bt + boff + 8);
                    acc1 = __builtin_amdgcn_wmma_f32_16x16x32_bf16(
                            false, A.v, false, Bf.v, (short)0, acc1, false, false);
                }
            }

            // ---- running argmin over this vtile's 16 columns ----
            #pragma unroll
            for (int r = 0; r < 8; ++r) {
                float dist = cbs - 2.0f * (acc0[r] + acc1[r]);
                unsigned int u = __float_as_uint(dist);
                u = (u & 0x80000000u) ? ~u : (u | 0x80000000u);  // orderable key
                unsigned long long key =
                    ((unsigned long long)u << 32) | (unsigned int)vcode;
                rmin[r] = key < rmin[r] ? key : rmin[r];
            }
            __syncthreads();   // everyone done reading bt; it may be overwritten
        }

        // ---- cross-lane / cross-wave reduction via ds_min_u64 ----
        {
            int rbase = mSub * 16 + ((lane >> 4) ? 8 : 0);  // C layout: lanes16-31 -> M+8
            #pragma unroll
            for (int r = 0; r < 8; ++r)
                atomicMin(&minPack[rbase + r], rmin[r]);
        }
        __syncthreads();

        if (tid < MTILE) {
            int v = (int)(unsigned int)(minPack[tid] & 0xFFFFFFFFull);
            tokHist[step][tid] = v;
            int gRow = rowBase + tid;
            int b = gRow >> 12;          // L = 4096
            int l = gRow & 4095;
            tokOut[((b * NCB + step) << 12) | l] = v;
        }
        __syncthreads();
    }

    // ---- quantized_sum = sum_s cb[s][tok_s] ----
    for (int i = tid; i < MTILE * (DDIM / 4); i += 256) {
        int row = i >> 7;
        int c4  = i & 127;
        float4 q; q.x = 0.f; q.y = 0.f; q.z = 0.f; q.w = 0.f;
        #pragma unroll
        for (int s = 0; s < NCB; ++s) {
            const float4 qq = ((const float4*)cb)[
                ((size_t)(s * VCODES + tokHist[s][row])) * 128 + c4];
            q.x += qq.x; q.y += qq.y; q.z += qq.z; q.w += qq.w;
        }
        ((float4*)qOut)[(size_t)(rowBase + row) * 128 + c4] = q;
    }
}

// ---------------------------------------------------------------------------
extern "C" void kernel_launch(void* const* d_in, const int* in_sizes, int n_in,
                              void* d_out, int out_size, void* d_ws, size_t ws_size,
                              hipStream_t stream) {
    const float* feat = (const float*)d_in[0];            // (8,4096,512) f32
    const float* cb   = (const float*)d_in[1];            // (4,4096,512) f32

    unsigned short* packedB = (unsigned short*)d_ws;      // 16 MB bf16 packed
    float* cbsq = (float*)((char*)d_ws +
                           (size_t)NCB * VCODES * DDIM * sizeof(unsigned short));

    int*   tokOut = (int*)d_out;                          // first 131072 i32
    float* qOut   = (float*)d_out + TOK_ELEMS;            // then (8,4096,512) f32

    // pre-pass: pack codebooks (bf16 fragment order) + squared norms
    rvq_pack_cb<<<(NCB * VCODES * DDIM) / 256, 256, 0, stream>>>(cb, packedB);
    rvq_cbsq<<<(NCB * VCODES) / 8, 256, 0, stream>>>(cb, cbsq);

    // fused 4-step RVQ
    rvq_main<<<ROWS / MTILE, 256, 0, stream>>>(feat, cb, packedB, cbsq,
                                               tokOut, qOut);
}
